// MultiHeadLatentAttention_85383949845162
// MI455X (gfx1250) — matrix operand order repaired and checked by reading, not compiled, not run on hardware
//
#include <hip/hip_runtime.h>
#include <hip/hip_bf16.h>
#include <stdint.h>

#define BB 4
#define SS 2048
#define DD 512
#define HH 8
#define LL 128
#define HD 64

typedef __attribute__((ext_vector_type(16))) __bf16 v16bf;
typedef __attribute__((ext_vector_type(8)))  float  v8f;

union FragU { uint4 u[2]; v16bf v; unsigned short s[16]; };

__device__ __forceinline__ unsigned short f2bf(float f) {
    unsigned int u = __float_as_uint(f);
    unsigned int r = u + 0x7FFFu + ((u >> 16) & 1u);
    return (unsigned short)(r >> 16);
}

// A-fragment (16x32 bf16, row-major source, ld = K):
// lane L holds row = L&15; lanes<16: K runs {k..k+7, k+16..k+23}; lanes>=16: {k+8..k+15, k+24..k+31}
__device__ __forceinline__ v16bf load_a_bf16(const unsigned short* A, long long row,
                                             int ld, int kBase, int lane) {
    int khalf = (lane >> 4) * 8;
    const unsigned short* p = A + row * (long long)ld + kBase + khalf;
    FragU f;
    f.u[0] = *(const uint4*)(p);
    f.u[1] = *(const uint4*)(p + 16);
    return f.v;
}

// B-fragment (32x16 bf16) from B^T layout (N x K row-major, ld = K):
// lane holds col = lane&15; lanes<16: K runs {k..k+7, k+16..k+23}; lanes>=16: {k+8..k+15, k+24..k+31}
__device__ __forceinline__ v16bf load_bt_bf16(const unsigned short* BT, long long col,
                                              int ld, int kBase, int lane) {
    int k0 = kBase + ((lane >> 4) * 8);
    const unsigned short* p = BT + col * (long long)ld + k0;
    FragU f;
    f.u[0] = *(const uint4*)(p);
    f.u[1] = *(const uint4*)(p + 16);
    return f.v;
}

// A-fragment from fp32 source (convert to bf16 in regs)
__device__ __forceinline__ v16bf load_a_f32(const float* A, long long row,
                                            int ld, int kBase, int lane) {
    int khalf = (lane >> 4) * 8;
    const float* p = A + row * (long long)ld + kBase + khalf;
    float4 x0 = *(const float4*)(p);
    float4 x1 = *(const float4*)(p + 4);
    float4 x2 = *(const float4*)(p + 16);
    float4 x3 = *(const float4*)(p + 20);
    FragU f;
    f.s[0]  = f2bf(x0.x); f.s[1]  = f2bf(x0.y); f.s[2]  = f2bf(x0.z); f.s[3]  = f2bf(x0.w);
    f.s[4]  = f2bf(x1.x); f.s[5]  = f2bf(x1.y); f.s[6]  = f2bf(x1.z); f.s[7]  = f2bf(x1.w);
    f.s[8]  = f2bf(x2.x); f.s[9]  = f2bf(x2.y); f.s[10] = f2bf(x2.z); f.s[11] = f2bf(x2.w);
    f.s[12] = f2bf(x3.x); f.s[13] = f2bf(x3.y); f.s[14] = f2bf(x3.z); f.s[15] = f2bf(x3.w);
    return f.v;
}

#define WMMA_BF16(A_, B_, C_) \
    __builtin_amdgcn_wmma_f32_16x16x32_bf16(false, (A_), false, (B_), (short)0, (C_), false, false)

// ---------------- elementwise converters ----------------
__global__ void cvt_bf16_kernel(const float* __restrict__ src, unsigned short* __restrict__ dst,
                                long long n) {
    long long i = (long long)blockIdx.x * blockDim.x + threadIdx.x;
    if (i < n) dst[i] = f2bf(src[i]);
}

__global__ void transpose_bf16_kernel(const float* __restrict__ src, unsigned short* __restrict__ dst,
                                      int R, int C) {
    long long i = (long long)blockIdx.x * blockDim.x + threadIdx.x;
    if (i < (long long)R * C) {
        int r = (int)(i / C), c = (int)(i % C);
        dst[(long long)c * R + r] = f2bf(src[i]);
    }
}

// ---------------- generic bf16 WMMA GEMM, 32x64 tile per wave ----------------
// mode 0: out bf16 = relu(acc+bias)              (latent)
// mode 1: out f32  = sigmoid(acc+bias)           (mod_q / mod_k)
// mode 2: out bf16 = (acc+bias)*mod[row,col]     (modulated q/k projection)
// mode 3: out bf16 scattered to vT[B,H,HD,S]     (v projection, transposed per head)
// mode 4: out f32  = acc+bias                    (final O projection)
__global__ void gemm_bf16_kernel(const unsigned short* __restrict__ A,
                                 const unsigned short* __restrict__ BT,
                                 const float* __restrict__ bias,
                                 const float* __restrict__ mod,
                                 float* __restrict__ outF,
                                 unsigned short* __restrict__ outB,
                                 int M, int N, int K, int mode) {
    int lane = threadIdx.x & 31;
    int wid  = threadIdx.x >> 5;
    int ntN  = N >> 6;                       // 64-wide col tiles
    int gw   = blockIdx.x * 8 + wid;
    if (gw >= (M >> 5) * ntN) return;
    int tM = gw / ntN, tN = gw % ntN;
    long long rowBase = (long long)tM * 32;
    int colBase = tN * 64;
    long long aRow0 = rowBase + (lane & 15);
    long long aRow1 = aRow0 + 16;

    v8f c0[4] = {{0}, {0}, {0}, {0}};
    v8f c1[4] = {{0}, {0}, {0}, {0}};
    for (int k = 0; k < K; k += 32) {
        v16bf a0 = load_a_bf16(A, aRow0, K, k, lane);
        v16bf a1 = load_a_bf16(A, aRow1, K, k, lane);
        #pragma unroll
        for (int j = 0; j < 4; ++j) {
            v16bf b = load_bt_bf16(BT, colBase + j * 16 + (lane & 15), K, k, lane);
            c0[j] = WMMA_BF16(a0, b, c0[j]);
            c1[j] = WMMA_BF16(a1, b, c1[j]);
        }
    }

    int n  = lane & 15;
    int hi = (lane >> 4) * 8;
    #pragma unroll
    for (int j = 0; j < 4; ++j) {
        int col = colBase + j * 16 + n;
        float bv = bias ? bias[col] : 0.0f;
        #pragma unroll
        for (int mi = 0; mi < 2; ++mi) {
            #pragma unroll
            for (int r = 0; r < 8; ++r) {
                long long row = rowBase + mi * 16 + hi + r;
                float v = (mi == 0 ? c0[j][r] : c1[j][r]) + bv;
                if (mode == 0) {
                    v = v > 0.0f ? v : 0.0f;
                    outB[row * N + col] = f2bf(v);
                } else if (mode == 1) {
                    outF[row * N + col] = 1.0f / (1.0f + __expf(-v));
                } else if (mode == 2) {
                    v *= mod[row * N + col];
                    outB[row * N + col] = f2bf(v);
                } else if (mode == 3) {
                    int b_ = (int)(row >> 11), s_ = (int)(row & (SS - 1));
                    int h_ = col >> 6, hd_ = col & (HD - 1);
                    outB[(((long long)(b_ * HH + h_) * HD + hd_) << 11) + s_] = f2bf(v);
                } else {
                    outF[row * N + col] = v;
                }
            }
        }
    }
}

// ---------------- attention pass 1: scores + online row max/sum ----------------
// 8 waves/block, all on the same (b,h); key tiles staged into LDS with async copies.
__global__ void attn_scores_kernel(const unsigned short* __restrict__ qp,
                                   const unsigned short* __restrict__ kp,
                                   float* __restrict__ sc,
                                   float* __restrict__ rowM,
                                   float* __restrict__ rowL) {
    __shared__ unsigned short shK[64 * 64];   // 64 key rows x 64 head-dim bf16 (8 KB)
    int lane = threadIdx.x & 31;
    int wid  = threadIdx.x >> 5;
    int tid  = threadIdx.x;
    int gw   = blockIdx.x * 8 + wid;          // 0 .. B*H*(S/16)-1 = 4095
    int bh   = gw >> 7;                        // same for all 8 waves of a block
    int rb   = gw & 127;
    int b = bh >> 3, h = bh & 7;
    const unsigned short* qh = qp + (long long)b * SS * DD + h * HD;
    const unsigned short* kh = kp + (long long)b * SS * DD + h * HD;
    long long aRow = rb * 16 + (lane & 15);
    v16bf a0 = load_a_bf16(qh, aRow, DD, 0, lane);   // reused across all column chunks
    v16bf a1 = load_a_bf16(qh, aRow, DD, 32, lane);

    float mrun[8], lrun[8];
    #pragma unroll
    for (int r = 0; r < 8; ++r) { mrun[r] = -1e30f; lrun[r] = 0.0f; }

    const float scale = 0.125f;  // 1/sqrt(64)
    long long scBase = ((long long)bh * SS + rb * 16) * SS;
    int n  = lane & 15;
    int hi = (lane >> 4) * 8;
    unsigned shBase = (unsigned)(uintptr_t)(&shK[0]);   // flat LDS addr truncates to LDS offset

    for (int cc = 0; cc < 32; ++cc) {          // 32 chunks of 64 key rows
        // cooperative async copy: 512 x 16B segments, 2 per thread
        #pragma unroll
        for (int seg = 0; seg < 2; ++seg) {
            int linear = tid + seg * 256;      // 0..511
            int krow = linear >> 3;            // 0..63
            int s8   = linear & 7;             // 16B segment within 128B row
            unsigned ldsAddr = shBase + (unsigned)(linear * 16);
            const unsigned short* g = kh + (long long)(cc * 64 + krow) * DD + s8 * 8;
            asm volatile("global_load_async_to_lds_b128 %0, %1, off"
                         :: "v"(ldsAddr), "v"(g) : "memory");
        }
        asm volatile("s_wait_asynccnt 0" ::: "memory");
        __syncthreads();

        #pragma unroll
        for (int sub = 0; sub < 4; ++sub) {
            const unsigned short* pb = &shK[(sub * 16 + (lane & 15)) * 64];
            int k0 = (lane >> 4) * 8;
            FragU f0, f1;
            f0.u[0] = *(const uint4*)(pb + k0);
            f0.u[1] = *(const uint4*)(pb + k0 + 16);
            f1.u[0] = *(const uint4*)(pb + 32 + k0);
            f1.u[1] = *(const uint4*)(pb + 32 + k0 + 16);
            v8f c = {};
            c = WMMA_BF16(a0, f0.v, c);
            c = WMMA_BF16(a1, f1.v, c);
            int colB = cc * 64 + sub * 16;
            #pragma unroll
            for (int r = 0; r < 8; ++r) {
                float v = c[r] * scale;
                sc[scBase + (long long)(hi + r) * SS + colB + n] = v;
                float tm = v;
                tm = fmaxf(tm, __shfl_xor(tm, 1, 16));
                tm = fmaxf(tm, __shfl_xor(tm, 2, 16));
                tm = fmaxf(tm, __shfl_xor(tm, 4, 16));
                tm = fmaxf(tm, __shfl_xor(tm, 8, 16));
                float nm = fmaxf(mrun[r], tm);
                float e = __expf(v - nm);
                e += __shfl_xor(e, 1, 16);
                e += __shfl_xor(e, 2, 16);
                e += __shfl_xor(e, 4, 16);
                e += __shfl_xor(e, 8, 16);
                lrun[r] = lrun[r] * __expf(mrun[r] - nm) + e;
                mrun[r] = nm;
            }
        }
        __syncthreads();
    }
    if ((lane & 15) == 0) {
        long long base = (long long)bh * SS + rb * 16 + hi;
        #pragma unroll
        for (int r = 0; r < 8; ++r) { rowM[base + r] = mrun[r]; rowL[base + r] = lrun[r]; }
    }
}

// ---------------- attention pass 2: in-place normalize ----------------
__global__ void softmax_norm_kernel(float* __restrict__ sc,
                                    const float* __restrict__ rowM,
                                    const float* __restrict__ rowL) {
    long long idx = (long long)blockIdx.x * blockDim.x + threadIdx.x;
    long long rowIdx = idx >> 11;    // / S
    float m = rowM[rowIdx], l = rowL[rowIdx];
    sc[idx] = __expf(sc[idx] - m) / l;
}

// ---------------- attention pass 3: attn_out = P @ V (16x64 per wave) ----------------
__global__ void attn_av_kernel(const float* __restrict__ P,
                               const unsigned short* __restrict__ vT,
                               unsigned short* __restrict__ ao) {
    int lane = threadIdx.x & 31;
    int wid  = threadIdx.x >> 5;
    int gw   = blockIdx.x * 8 + wid;          // 0 .. 4095
    int bh   = gw >> 7;
    int tM   = gw & 127;
    int b = bh >> 3, h = bh & 7;
    const float* Ph = P + (long long)bh * SS * SS;
    const unsigned short* vh = vT + (long long)bh * HD * SS;
    long long aRow = tM * 16 + (lane & 15);
    v8f c[4] = {{0}, {0}, {0}, {0}};
    for (int k = 0; k < SS; k += 32) {
        v16bf a = load_a_f32(Ph, aRow, SS, k, lane);
        #pragma unroll
        for (int j = 0; j < 4; ++j) {
            v16bf bb = load_bt_bf16(vh, j * 16 + (lane & 15), SS, k, lane);
            c[j] = WMMA_BF16(a, bb, c[j]);
        }
    }
    int n = lane & 15, hi = (lane >> 4) * 8;
    #pragma unroll
    for (int j = 0; j < 4; ++j) {
        #pragma unroll
        for (int r = 0; r < 8; ++r) {
            long long srow = (long long)b * SS + tM * 16 + hi + r;
            ao[srow * DD + h * HD + j * 16 + n] = f2bf(c[j][r]);
        }
    }
}

extern "C" void kernel_launch(void* const* d_in, const int* in_sizes, int n_in,
                              void* d_out, int out_size, void* d_ws, size_t ws_size,
                              hipStream_t stream) {
    const float* query = (const float*)d_in[0];
    const float* key   = (const float*)d_in[1];
    const float* value = (const float*)d_in[2];
    const float* Wq = (const float*)d_in[3];  const float* bq = (const float*)d_in[4];
    const float* Wk = (const float*)d_in[5];  const float* bk = (const float*)d_in[6];
    const float* Wv = (const float*)d_in[7];  const float* bv = (const float*)d_in[8];
    const float* Wo = (const float*)d_in[9];  const float* bo = (const float*)d_in[10];
    const float* Wg = (const float*)d_in[11]; const float* bg = (const float*)d_in[12];
    const float* Wmq = (const float*)d_in[13]; const float* bmq = (const float*)d_in[14];
    const float* Wmk = (const float*)d_in[15]; const float* bmk = (const float*)d_in[16];

    float* outO  = (float*)d_out;                       // [B,S,D] = 4194304 f32
    float* attnW = outO + (long long)BB * SS * DD;      // [B,H,S,S] = 134217728 f32

    const long long MQ = (long long)BB * SS;            // 8192 rows

    // workspace carve (256B aligned)
    char* w = (char*)d_ws;
    auto carve = [&](size_t bytes) -> void* {
        void* p = (void*)w;
        w += (bytes + 255) & ~(size_t)255;
        return p;
    };
    unsigned short* qb   = (unsigned short*)carve(MQ * DD * 2);
    unsigned short* kb   = (unsigned short*)carve(MQ * DD * 2);
    unsigned short* vb   = (unsigned short*)carve(MQ * DD * 2);
    unsigned short* WqT  = (unsigned short*)carve((size_t)DD * DD * 2);
    unsigned short* WkT  = (unsigned short*)carve((size_t)DD * DD * 2);
    unsigned short* WvT  = (unsigned short*)carve((size_t)DD * DD * 2);
    unsigned short* WoT  = (unsigned short*)carve((size_t)DD * DD * 2);
    unsigned short* WgT  = (unsigned short*)carve((size_t)LL * DD * 2);
    unsigned short* WmqT = (unsigned short*)carve((size_t)DD * LL * 2);
    unsigned short* WmkT = (unsigned short*)carve((size_t)DD * LL * 2);
    unsigned short* latb = (unsigned short*)carve(MQ * LL * 2);
    float* modq          = (float*)carve(MQ * DD * 4);
    float* modk          = (float*)carve(MQ * DD * 4);
    unsigned short* qp   = (unsigned short*)carve(MQ * DD * 2);
    unsigned short* kp   = (unsigned short*)carve(MQ * DD * 2);
    unsigned short* vT   = (unsigned short*)carve(MQ * DD * 2);
    unsigned short* aob  = (unsigned short*)carve(MQ * DD * 2);
    float* rowM          = (float*)carve((size_t)BB * HH * SS * 4);
    float* rowL          = (float*)carve((size_t)BB * HH * SS * 4);

    const int T = 256;
    long long nAct = MQ * DD;

    // 1) convert activations + transpose weights to bf16
    cvt_bf16_kernel<<<(int)((nAct + T - 1) / T), T, 0, stream>>>(query, qb, nAct);
    cvt_bf16_kernel<<<(int)((nAct + T - 1) / T), T, 0, stream>>>(key,   kb, nAct);
    cvt_bf16_kernel<<<(int)((nAct + T - 1) / T), T, 0, stream>>>(value, vb, nAct);
    transpose_bf16_kernel<<<(DD * DD + T - 1) / T, T, 0, stream>>>(Wq, WqT, DD, DD);
    transpose_bf16_kernel<<<(DD * DD + T - 1) / T, T, 0, stream>>>(Wk, WkT, DD, DD);
    transpose_bf16_kernel<<<(DD * DD + T - 1) / T, T, 0, stream>>>(Wv, WvT, DD, DD);
    transpose_bf16_kernel<<<(DD * DD + T - 1) / T, T, 0, stream>>>(Wo, WoT, DD, DD);
    transpose_bf16_kernel<<<(DD * LL + T - 1) / T, T, 0, stream>>>(Wg, WgT, DD, LL);
    transpose_bf16_kernel<<<(LL * DD + T - 1) / T, T, 0, stream>>>(Wmq, WmqT, LL, DD);
    transpose_bf16_kernel<<<(LL * DD + T - 1) / T, T, 0, stream>>>(Wmk, WmkT, LL, DD);

    // 2) latent = relu(query @ Wg + bg)        M=8192 N=128 K=512
    {
        int tiles = (int)(MQ / 32) * (LL / 64);
        gemm_bf16_kernel<<<tiles / 8, T, 0, stream>>>(qb, WgT, bg, nullptr, nullptr, latb,
                                                      (int)MQ, LL, DD, 0);
    }
    // 3) mod_q / mod_k = sigmoid(latent @ Wm* + b)   M=8192 N=512 K=128
    {
        int tiles = (int)(MQ / 32) * (DD / 64);
        gemm_bf16_kernel<<<tiles / 8, T, 0, stream>>>(latb, WmqT, bmq, nullptr, modq, nullptr,
                                                      (int)MQ, DD, LL, 1);
        gemm_bf16_kernel<<<tiles / 8, T, 0, stream>>>(latb, WmkT, bmk, nullptr, modk, nullptr,
                                                      (int)MQ, DD, LL, 1);
    }
    // 4) q/k/v projections   M=8192 N=512 K=512
    {
        int tiles = (int)(MQ / 32) * (DD / 64);
        gemm_bf16_kernel<<<tiles / 8, T, 0, stream>>>(qb, WqT, bq, modq, nullptr, qp,
                                                      (int)MQ, DD, DD, 2);
        gemm_bf16_kernel<<<tiles / 8, T, 0, stream>>>(kb, WkT, bk, modk, nullptr, kp,
                                                      (int)MQ, DD, DD, 2);
        gemm_bf16_kernel<<<tiles / 8, T, 0, stream>>>(vb, WvT, bv, nullptr, nullptr, vT,
                                                      (int)MQ, DD, DD, 3);
    }
    // 5) scores + softmax stats (raw scaled scores into d_out attn region)
    attn_scores_kernel<<<(BB * HH * (SS / 16)) / 8, T, 0, stream>>>(qp, kp, attnW, rowM, rowL);
    // 6) normalize in place
    {
        long long nw = (long long)BB * HH * SS * SS;
        softmax_norm_kernel<<<(int)(nw / T), T, 0, stream>>>(attnW, rowM, rowL);
    }
    // 7) attn_out = P @ V   (per head M=2048 N=64 K=2048; 16x64 tile per wave)
    attn_av_kernel<<<(BB * HH * (SS / 16)) / 8, T, 0, stream>>>(attnW, vT, aob);
    // 8) output = attn_out @ Wo + bo   M=8192 N=512 K=512
    {
        int tiles = (int)(MQ / 32) * (DD / 64);
        gemm_bf16_kernel<<<tiles / 8, T, 0, stream>>>(aob, WoT, bo, nullptr, outO, nullptr,
                                                      (int)MQ, DD, DD, 4);
    }
}
